// Megatron_MHA_82386062672028
// MI455X (gfx1250) — compile-verified, matrix operand order checked by reading, not run on hardware
//
#include <hip/hip_runtime.h>

// ---------------------------------------------------------------------------
// Megatron-style causal MHA forward for MI455X (gfx1250, wave32, WMMA).
//   qkv  = x @ Wqkv^T                    (bf16 WMMA, f32 accum)
//   attn = causal softmax(q k^T / 8) v   (flash-attention, bf16 WMMA)
//   out  = attn @ Wproj^T                (bf16 WMMA, f32 out)
// Round 3: GEMM block 128x128, each wave 32x64 = 8 accumulators so each
// 32-wide k-step is 12 ds_load_b128 -> 8 WMMAs (B-frags reused across two
// A-frags). Double-buffered LDS, one barrier per 64-wide K-step, prefetch.
// ---------------------------------------------------------------------------

typedef __attribute__((ext_vector_type(16))) __bf16 v16bf;
typedef __attribute__((ext_vector_type(8)))  float  v8f;

#define NHEAD 16
#define HDIM  64
#define NEMBD 1024
#define BATCH 4
#define SEQ   2048
#define MTOT  (BATCH * SEQ)    // 8192 rows
#define QKVN  (3 * NEMBD)      // 3072

// fp32 -> bf16 round-to-nearest-even (finite inputs)
__device__ __forceinline__ unsigned short f2bf(float f) {
  union { float f; unsigned int u; } v; v.f = f;
  unsigned int u = v.u;
  u += 0x7FFFu + ((u >> 16) & 1u);
  return (unsigned short)(u >> 16);
}

// Load one 16x32 bf16 A/B fragment from LDS (row-major, ld in elements,
// ld must be a multiple of 8 so every b128 access is 16B aligned).
// Per CDNA5 ISA 7.12.2: lane -> row lane&15; half = lane>>4 holds
// K = half*8..+7 in dwords 0..3 and K = 16+half*8..+7 in dwords 4..7.
__device__ __forceinline__ v16bf load_frag(const unsigned short* base, int lane, int ld) {
  const unsigned short* p = base + (lane & 15) * ld + ((lane >> 4) << 3);
  union { uint4 q[2]; v16bf v; } u;
  u.q[0] = *(const uint4*)(p);
  u.q[1] = *(const uint4*)(p + 16);
  return u.v;
}

// ----------------------------- cast kernel ---------------------------------
__global__ __launch_bounds__(256) void cast_bf16(const float4* __restrict__ in,
                                                 uint2* __restrict__ out, int n4) {
  int i = blockIdx.x * blockDim.x + threadIdx.x;
  int stride = gridDim.x * blockDim.x;
  for (; i < n4; i += stride) {
    float4 f = in[i];
    uint2 o;
    o.x = (unsigned int)f2bf(f.x) | ((unsigned int)f2bf(f.y) << 16);
    o.y = (unsigned int)f2bf(f.z) | ((unsigned int)f2bf(f.w) << 16);
    out[i] = o;
  }
}

// ----------------------------- NT GEMM -------------------------------------
// C[M,N] = A[M,K] * B[N,K]^T ; A,B bf16 row-major; C fp32 or bf16.
// Block tile 128(M) x 128(N), K step 64, double-buffered LDS. 256 threads =
// 8 waves as 4(M)x2(N); each wave owns a 32x64 slab = 8 f32 accumulators.
#define BM 128
#define BN 128
#define BK 64
#define LDA 72   // BK + 8 pad (multiple of 8 -> 16B-aligned rows, bank rotate)

// Stage one 128x64 bf16 tile of A and of B into LDS (32 elems per thread each).
__device__ __forceinline__ void stage_tile(const unsigned short* __restrict__ A,
                                           const unsigned short* __restrict__ B,
                                           unsigned short* As, unsigned short* Bs,
                                           int bm, int bn, int K, int k0,
                                           int grow, int gcol) {
  const unsigned short* ap = &A[(size_t)(bm + grow) * K + k0 + gcol];
#pragma unroll
  for (int i = 0; i < 4; ++i)
    *(uint4*)&As[grow * LDA + gcol + i * 8] = *(const uint4*)(ap + i * 8);
  const unsigned short* bp = &B[(size_t)(bn + grow) * K + k0 + gcol];
#pragma unroll
  for (int i = 0; i < 4; ++i)
    *(uint4*)&Bs[grow * LDA + gcol + i * 8] = *(const uint4*)(bp + i * 8);
}

__global__ __launch_bounds__(256) void gemm_nt(const unsigned short* __restrict__ A,
                                               const unsigned short* __restrict__ B,
                                               void* __restrict__ C,
                                               int N, int K, int out_bf16) {
  __shared__ unsigned short As[2][BM * LDA];
  __shared__ unsigned short Bs[2][BN * LDA];
  const int tid  = threadIdx.x;
  const int lane = tid & 31;
  const int wave = tid >> 5;
  const int bm = blockIdx.x * BM;
  const int bn = blockIdx.y * BN;
  const int wm = (wave & 3) * 32;   // wave M offset in block (2 x 16-row tiles)
  const int wn = (wave >> 2) * 64;  // wave N offset in block (4 x 16-col tiles)
  const int grow = tid >> 1, gcol = (tid & 1) * 32;  // 128x64 staging map

  v8f acc[2][4];
#pragma unroll
  for (int dm = 0; dm < 2; ++dm)
#pragma unroll
    for (int j = 0; j < 4; ++j)
#pragma unroll
      for (int e = 0; e < 8; ++e) acc[dm][j][e] = 0.f;

  // prologue: stage tile 0 into buffer 0
  stage_tile(A, B, As[0], Bs[0], bm, bn, K, 0, grow, gcol);
  __syncthreads();

  int p = 0;
  for (int k0 = 0; k0 < K; k0 += BK) {
    // overlap: stage next tile into the other buffer while computing this one
    if (k0 + BK < K)
      stage_tile(A, B, As[p ^ 1], Bs[p ^ 1], bm, bn, K, k0 + BK, grow, gcol);
    // prefetch the tile after next into GL2 (global_prefetch_b8)
    if (k0 + 2 * BK < K) {
      __builtin_prefetch(&A[(size_t)(bm + grow) * K + k0 + 2 * BK + gcol], 0, 1);
      __builtin_prefetch(&B[(size_t)(bn + grow) * K + k0 + 2 * BK + gcol], 0, 1);
    }

    // per 32-wide k-step: 6 fragment loads (12 ds_load_b128) -> 8 WMMAs
#pragma unroll
    for (int dk = 0; dk < 2; ++dk) {
      v16bf af[2], bf[4];
#pragma unroll
      for (int dm = 0; dm < 2; ++dm)
        af[dm] = load_frag(&As[p][(wm + dm * 16) * LDA + dk * 32], lane, LDA);
#pragma unroll
      for (int j = 0; j < 4; ++j)
        bf[j] = load_frag(&Bs[p][(wn + j * 16) * LDA + dk * 32], lane, LDA);
#pragma unroll
      for (int dm = 0; dm < 2; ++dm)
#pragma unroll
        for (int j = 0; j < 4; ++j)
          acc[dm][j] = __builtin_amdgcn_wmma_f32_16x16x32_bf16(
              false, af[dm], false, bf[j], (short)0, acc[dm][j], false, false);
    }
    __syncthreads();  // one barrier per K-step: compute(p) done, stores(p^1) done
    p ^= 1;
  }

  // C/D layout: vgpr r, lane -> row r + 8*(lane>>4), col lane&15
  const int col = lane & 15, half = lane >> 4;
  if (out_bf16) {
    unsigned short* Cb = (unsigned short*)C;
#pragma unroll
    for (int dm = 0; dm < 2; ++dm)
#pragma unroll
      for (int j = 0; j < 4; ++j)
#pragma unroll
        for (int r = 0; r < 8; ++r)
          Cb[(size_t)(bm + wm + dm * 16 + r + 8 * half) * N + (bn + wn + j * 16 + col)] =
              f2bf(acc[dm][j][r]);
  } else {
    float* Cf = (float*)C;
#pragma unroll
    for (int dm = 0; dm < 2; ++dm)
#pragma unroll
      for (int j = 0; j < 4; ++j)
#pragma unroll
        for (int r = 0; r < 8; ++r)
          Cf[(size_t)(bm + wm + dm * 16 + r + 8 * half) * N + (bn + wn + j * 16 + col)] =
              acc[dm][j][r];
  }
}

// ----------------------- flash attention (causal) --------------------------
// Grid: (SEQ/64, BATCH*NHEAD). 128 threads = 4 waves; wave w owns 16 query
// rows. Per 32-key block: S = Q K^T (WMMA), online softmax (f32), P -> LDS
// (C-layout -> A-layout bounce), O += P V (WMMA, V staged transposed in LDS).
#define AQ  64
#define AKB 32
#define QLD 72   // HDIM + 8 pad
#define VLD 40   // AKB + 8 pad

__global__ __launch_bounds__(128) void attn_fwd(const unsigned short* __restrict__ qkv,
                                                unsigned short* __restrict__ outb) {
  __shared__ unsigned short Qs[AQ * QLD];        // [qrow][d]
  __shared__ unsigned short Ks[AKB * QLD];       // [krow][d]
  __shared__ unsigned short Vt[HDIM * VLD];      // [d][krow]  (transposed)
  __shared__ unsigned short Ps[4 * 16 * VLD];    // per-wave P tile [m][krow]

  const int tid = threadIdx.x, lane = tid & 31, wave = tid >> 5;
  const int q0 = blockIdx.x * AQ;
  const int b  = blockIdx.y >> 4;
  const int h  = blockIdx.y & 15;
  const size_t rowbase = (size_t)(b * SEQ) * QKVN;
  const int hc = h * HDIM;
  const int col = lane & 15, half = lane >> 4;
  const int wq = wave * 16;

  // stage Q tile once
  {
    int r = tid >> 1, cb = (tid & 1) * 32;
    const unsigned short* src = qkv + rowbase + (size_t)(q0 + r) * QKVN + hc + cb;
#pragma unroll
    for (int i = 0; i < 4; ++i)
      *(uint4*)&Qs[r * QLD + cb + i * 8] = *(const uint4*)(src + i * 8);
  }

  float m_st[8], l_st[8];
  v8f acc_o[4];
#pragma unroll
  for (int r = 0; r < 8; ++r) { m_st[r] = -1e30f; l_st[r] = 0.f; }
#pragma unroll
  for (int j = 0; j < 4; ++j)
#pragma unroll
    for (int e = 0; e < 8; ++e) acc_o[j][e] = 0.f;

  const int kend = q0 + AQ;  // uniform across the workgroup
  for (int k0 = 0; k0 < kend; k0 += AKB) {
    __syncthreads();
    // stage K block (row-major)
    {
      int r = tid >> 2, cb = (tid & 3) * 16;
      const unsigned short* src = qkv + rowbase + (size_t)(k0 + r) * QKVN + NEMBD + hc + cb;
      *(uint4*)&Ks[r * QLD + cb]     = *(const uint4*)(src);
      *(uint4*)&Ks[r * QLD + cb + 8] = *(const uint4*)(src + 8);
    }
    // stage V block transposed: Vt[d][krow]
    {
      int r = tid >> 2, db = (tid & 3) * 16;
      const unsigned short* src = qkv + rowbase + (size_t)(k0 + r) * QKVN + 2 * NEMBD + hc + db;
      union { uint4 q; unsigned short s[8]; } u0, u1;
      u0.q = *(const uint4*)(src);
      u1.q = *(const uint4*)(src + 8);
#pragma unroll
      for (int e = 0; e < 8; ++e) {
        Vt[(db + e) * VLD + r]     = u0.s[e];
        Vt[(db + 8 + e) * VLD + r] = u1.s[e];
      }
    }
    __syncthreads();

    // S = Q K^T (16 q-rows x 32 keys): batch all 6 fragment loads, then
    // 4 WMMAs (D=64 -> two k=32 steps, accumulated)
    v16bf qa[2], kb[2][2];
#pragma unroll
    for (int dk = 0; dk < 2; ++dk) {
      qa[dk] = load_frag(&Qs[wq * QLD + dk * 32], lane, QLD);
#pragma unroll
      for (int c = 0; c < 2; ++c)
        kb[dk][c] = load_frag(&Ks[(c * 16) * QLD + dk * 32], lane, QLD);
    }
    v8f sacc[2];
#pragma unroll
    for (int c = 0; c < 2; ++c)
#pragma unroll
      for (int e = 0; e < 8; ++e) sacc[c][e] = 0.f;
#pragma unroll
    for (int dk = 0; dk < 2; ++dk)
#pragma unroll
      for (int c = 0; c < 2; ++c)
        sacc[c] = __builtin_amdgcn_wmma_f32_16x16x32_bf16(false, qa[dk], false, kb[dk][c],
                                                          (short)0, sacc[c], false, false);

    // --- online softmax (rows live across 16 lanes of a half-wave) ---
    float sv[2][8];
#pragma unroll
    for (int c = 0; c < 2; ++c)
#pragma unroll
      for (int r = 0; r < 8; ++r) {
        float s = sacc[c][r] * 0.125f;  // 1/sqrt(64)
        int kg = k0 + c * 16 + col;
        int qg = q0 + wq + r + 8 * half;
        sv[c][r] = (kg <= qg) ? s : -1e30f;
      }
    float al[8];
#pragma unroll
    for (int r = 0; r < 8; ++r) {
      float v = fmaxf(sv[0][r], sv[1][r]);
      v = fmaxf(v, __shfl_xor(v, 1, 32));
      v = fmaxf(v, __shfl_xor(v, 2, 32));
      v = fmaxf(v, __shfl_xor(v, 4, 32));
      v = fmaxf(v, __shfl_xor(v, 8, 32));
      float nm = fmaxf(m_st[r], v);
      al[r] = __expf(m_st[r] - nm);
      m_st[r] = nm;
    }
#pragma unroll
    for (int c = 0; c < 2; ++c)
#pragma unroll
      for (int r = 0; r < 8; ++r) sv[c][r] = __expf(sv[c][r] - m_st[r]);
#pragma unroll
    for (int r = 0; r < 8; ++r) {
      float s = sv[0][r] + sv[1][r];
      s += __shfl_xor(s, 1, 32);
      s += __shfl_xor(s, 2, 32);
      s += __shfl_xor(s, 4, 32);
      s += __shfl_xor(s, 8, 32);
      l_st[r] = l_st[r] * al[r] + s;
    }
#pragma unroll
    for (int j = 0; j < 4; ++j)
#pragma unroll
      for (int r = 0; r < 8; ++r) acc_o[j][r] *= al[r];

    // bounce P through LDS: C-layout -> row-major [m][krow]
#pragma unroll
    for (int c = 0; c < 2; ++c)
#pragma unroll
      for (int r = 0; r < 8; ++r)
        Ps[(wq + r + 8 * half) * VLD + c * 16 + col] = f2bf(sv[c][r]);
    __syncthreads();

    // O += P (16x32) x V (32x64): batch 5 fragment loads, then 4 WMMAs
    v16bf pa = load_frag(&Ps[wq * VLD], lane, VLD);
    v16bf vb[4];
#pragma unroll
    for (int j = 0; j < 4; ++j)
      vb[j] = load_frag(&Vt[(j * 16) * VLD], lane, VLD);
#pragma unroll
    for (int j = 0; j < 4; ++j)
      acc_o[j] = __builtin_amdgcn_wmma_f32_16x16x32_bf16(false, pa, false, vb[j],
                                                         (short)0, acc_o[j], false, false);
  }

  // epilogue: normalize and write [B*T, NEMBD] bf16
#pragma unroll
  for (int j = 0; j < 4; ++j)
#pragma unroll
    for (int r = 0; r < 8; ++r) {
      int m = b * SEQ + q0 + wq + r + 8 * half;
      outb[(size_t)m * NEMBD + hc + j * 16 + col] = f2bf(acc_o[j][r] / l_st[r]);
    }
}

// ----------------------------- launch --------------------------------------
extern "C" void kernel_launch(void* const* d_in, const int* in_sizes, int n_in,
                              void* d_out, int out_size, void* d_ws, size_t ws_size,
                              hipStream_t stream) {
  (void)in_sizes; (void)n_in; (void)out_size; (void)ws_size;
  const float* x     = (const float*)d_in[0];
  const float* Wqkv  = (const float*)d_in[1];
  const float* Wproj = (const float*)d_in[2];
  float* out = (float*)d_out;

  // workspace layout (bytes): xb 16MB | wqkvb 6MB | wprojb 2MB | qkvb 48MB
  char* ws = (char*)d_ws;
  const size_t xb_bytes    = (size_t)MTOT * NEMBD * 2;
  const size_t wqkv_bytes  = (size_t)QKVN * NEMBD * 2;
  const size_t wproj_bytes = (size_t)NEMBD * NEMBD * 2;
  unsigned short* xb     = (unsigned short*)(ws);
  unsigned short* wqkvb  = (unsigned short*)(ws + xb_bytes);
  unsigned short* wprojb = (unsigned short*)(ws + xb_bytes + wqkv_bytes);
  unsigned short* qkvb   = (unsigned short*)(ws + xb_bytes + wqkv_bytes + wproj_bytes);
  unsigned short* attnb  = xb;  // x is dead after the QKV GEMM; reuse its slot

  cast_bf16<<<dim3(1024), dim3(256), 0, stream>>>((const float4*)x, (uint2*)xb,
                                                  MTOT * NEMBD / 4);
  cast_bf16<<<dim3(768),  dim3(256), 0, stream>>>((const float4*)Wqkv, (uint2*)wqkvb,
                                                  QKVN * NEMBD / 4);
  cast_bf16<<<dim3(256),  dim3(256), 0, stream>>>((const float4*)Wproj, (uint2*)wprojb,
                                                  NEMBD * NEMBD / 4);

  // qkv = x @ Wqkv^T   [8192,1024] x [3072,1024]^T -> bf16 [8192,3072]
  gemm_nt<<<dim3(MTOT / BM, QKVN / BN), dim3(256), 0, stream>>>(xb, wqkvb, qkvb,
                                                                QKVN, NEMBD, 1);
  // causal flash attention -> bf16 [8192,1024]
  attn_fwd<<<dim3(SEQ / AQ, BATCH * NHEAD), dim3(128), 0, stream>>>(qkvb, attnb);
  // out = attn @ Wproj^T -> fp32 d_out
  gemm_nt<<<dim3(MTOT / BM, NEMBD / BN), dim3(256), 0, stream>>>(attnb, wprojb,
                                                                 (void*)out, NEMBD, NEMBD, 0);
}
